// RNN_43001212568048
// MI455X (gfx1250) — matrix operand order repaired and checked by reading, not compiled, not run on hardware
//
#include <hip/hip_runtime.h>

// Problem constants (from reference)
#define VOCAB  256
#define HDIM   1024
#define OUTD   256
#define NB     64
#define TSTEPS 1024

#define GLOBAL_AS __attribute__((address_space(1)))

typedef __attribute__((ext_vector_type(16))) __bf16       v16bf;
typedef __attribute__((ext_vector_type(8)))  float        v8f;
typedef __attribute__((ext_vector_type(4)))  unsigned int v4u;

union FragB16 {
    v16bf v;
    v4u   q[2];
};

__device__ __forceinline__ v4u ldg_b128(GLOBAL_AS const void* p) {
    return *(GLOBAL_AS const v4u*)p;
}
__device__ __forceinline__ v4u lds_b128(const void* p) {
    return *(const v4u*)p;
}
__device__ __forceinline__ float ldg_f32(const float* p) {
    return *(GLOBAL_AS const float*)p;
}

__device__ __forceinline__ float fast_tanh(float v) {
#if __has_builtin(__builtin_amdgcn_tanhf)
    return __builtin_amdgcn_tanhf(v);
#else
    return tanhf(v);
#endif
}

// ---------------------------------------------------------------------------
// Prep: transpose + bf16-pack the weights so WMMA B-fragments are contiguous.
//   Wpk[n*H + k]    = bf16( Whh[k*H + n] )      (H x H)
//   WoutPk[n*H + k] = bf16( Wout[k*OUT + n] )   (OUT x H)
// B-fragment as loaded here: lane n holds K = kb*32 + (lane>=16 ? 16 : 0) + 0..15,
// i.e. 16 consecutive bf16 -> two aligned 16B loads.
// ---------------------------------------------------------------------------
__global__ void prep_weights(const float* __restrict__ Whh,
                             const float* __restrict__ Wout,
                             __bf16* __restrict__ Wpk,
                             __bf16* __restrict__ WoutPk) {
    int idx = blockIdx.x * blockDim.x + threadIdx.x;
    if (idx < HDIM * HDIM) {
        int n = idx >> 10;            // row of packed layout
        int k = idx & (HDIM - 1);
        Wpk[idx] = (__bf16)Whh[k * HDIM + n];
    }
    if (idx < OUTD * HDIM) {
        int n = idx >> 10;
        int k = idx & (HDIM - 1);
        WoutPk[idx] = (__bf16)Wout[k * OUTD + n];
    }
}

// ---------------------------------------------------------------------------
// Fused RNN: 4 workgroups x 512 threads (16 wave32s). Each WG owns 16 samples,
// keeps h (16 x 1024) in LDS as bf16, loops over all T steps with workgroup
// barriers only (samples are independent -> no grid sync).
// ---------------------------------------------------------------------------
__launch_bounds__(512, 1)
__global__ void rnn_fused(const int*    __restrict__ x,
                          const float*  __restrict__ E,
                          const float*  __restrict__ bh,
                          const float*  __restrict__ bout,
                          const __bf16* __restrict__ Wpk,
                          const __bf16* __restrict__ WoutPk,
                          float* __restrict__ y,
                          float* __restrict__ hlast) {
    __shared__ __bf16 hbuf[16][HDIM];   // 32 KB: h_{t-1} -> h_t (bf16)
    __shared__ float  lg[16][OUTD];     // 16 KB: logits
    __shared__ int    tok[16];

    const int tid = threadIdx.x;
    const int wv  = tid >> 5;    // wave id 0..15
    const int l   = tid & 31;    // lane
    const int hi  = l >> 4;      // lane half (0/1)
    const int lm  = l & 15;      // lane-in-half
    const int rowBase = blockIdx.x * 16;

    // h_0 = 0
    for (int i = tid; i < 16 * HDIM; i += 512)
        ((__bf16*)hbuf)[i] = (__bf16)0.0f;
    __syncthreads();

    // Per-wave biases (loop-invariant, keep in regs).
    const float bh0 = bh[(wv * 4 + 0) * 16 + lm];
    const float bh1 = bh[(wv * 4 + 1) * 16 + lm];
    const float bh2 = bh[(wv * 4 + 2) * 16 + lm];
    const float bh3 = bh[(wv * 4 + 3) * 16 + lm];
    const float bo  = bout[wv * 16 + lm];

    // Per-lane byte offsets of the B-fragment rows inside the packed weights.
    const unsigned ro0 = (unsigned)((wv * 4 + 0) * 16 + lm) * (HDIM * 2);
    const unsigned ro1 = (unsigned)((wv * 4 + 1) * 16 + lm) * (HDIM * 2);
    const unsigned ro2 = (unsigned)((wv * 4 + 2) * 16 + lm) * (HDIM * 2);
    const unsigned ro3 = (unsigned)((wv * 4 + 3) * 16 + lm) * (HDIM * 2);
    const unsigned roo = (unsigned)(wv * 16 + lm) * (HDIM * 2);

    for (int t = 0; t < TSTEPS; ++t) {
        if (tid < 16) tok[tid] = x[(rowBase + tid) * TSTEPS + t];

        // Opaque *global-AS* weight bases: forbid LICM of the weight loads out
        // of the t-loop (round 1 spilled 4KB/lane to scratch) while keeping
        // address-space 1 so loads lower to global_load_b128 (round 2's
        // generic launder degraded them to flat_load, which also ties DScnt).
        GLOBAL_AS const char* Wg  = (GLOBAL_AS const char*)Wpk;
        GLOBAL_AS const char* Wog = (GLOBAL_AS const char*)WoutPk;
        asm volatile("" : "+s"(Wg), "+s"(Wog));

        // ---- stage 1: recurrent GEMM, 4 independent 16x16 tiles, K=1024 ----
        v8f acc0 = {}, acc1 = {}, acc2 = {}, acc3 = {};
        #pragma unroll 2
        for (int kb = 0; kb < HDIM / 32; ++kb) {
            FragB16 a;
            const int kA = kb * 32 + hi * 16;       // A layout: 16 contiguous K
            a.q[0] = lds_b128(&hbuf[lm][kA]);
            a.q[1] = lds_b128(&hbuf[lm][kA + 8]);
            const unsigned kB = (unsigned)(kb * 32 + hi * 16) * 2;  // bytes; 16 contiguous K
            FragB16 b0, b1, b2, b3;
            b0.q[0] = ldg_b128(Wg + ro0 + kB);  b0.q[1] = ldg_b128(Wg + ro0 + kB + 16);
            b1.q[0] = ldg_b128(Wg + ro1 + kB);  b1.q[1] = ldg_b128(Wg + ro1 + kB + 16);
            b2.q[0] = ldg_b128(Wg + ro2 + kB);  b2.q[1] = ldg_b128(Wg + ro2 + kB + 16);
            b3.q[0] = ldg_b128(Wg + ro3 + kB);  b3.q[1] = ldg_b128(Wg + ro3 + kB + 16);
            acc0 = __builtin_amdgcn_wmma_f32_16x16x32_bf16(false, a.v, false, b0.v, (short)0, acc0, false, false);
            acc1 = __builtin_amdgcn_wmma_f32_16x16x32_bf16(false, a.v, false, b1.v, (short)0, acc1, false, false);
            acc2 = __builtin_amdgcn_wmma_f32_16x16x32_bf16(false, a.v, false, b2.v, (short)0, acc2, false, false);
            acc3 = __builtin_amdgcn_wmma_f32_16x16x32_bf16(false, a.v, false, b3.v, (short)0, acc3, false, false);
        }
        __syncthreads();  // b1: all hbuf reads done; tok visible

        // ---- stage 3: + embedding + bias, tanh, write h_t back to LDS ----
        // Batched: 8 token reads, then 32 independent E gathers (one latency),
        // then pure VALU + LDS stores. No per-element branches.
        int tk[8];
        #pragma unroll
        for (int g = 0; g < 8; ++g) tk[g] = tok[g + 8 * hi];

        const float* erow[8];
        #pragma unroll
        for (int g = 0; g < 8; ++g) erow[g] = E + (size_t)tk[g] * HDIM;

        float ev[4][8];
        #pragma unroll
        for (int ci = 0; ci < 4; ++ci) {
            const int colB = (wv * 4 + ci) * 16 + lm;
            #pragma unroll
            for (int g = 0; g < 8; ++g)
                ev[ci][g] = ldg_f32(erow[g] + colB);
        }
        #pragma unroll
        for (int ci = 0; ci < 4; ++ci) {
            const int colB = (wv * 4 + ci) * 16 + lm;
            const v8f acc = (ci == 0) ? acc0 : (ci == 1) ? acc1 : (ci == 2) ? acc2 : acc3;
            const float bhv = (ci == 0) ? bh0 : (ci == 1) ? bh1 : (ci == 2) ? bh2 : bh3;
            #pragma unroll
            for (int g = 0; g < 8; ++g) {
                const float hv = fast_tanh(acc[g] + ev[ci][g] + bhv);
                ev[ci][g] = hv;                                  // keep f32 copy
                hbuf[g + 8 * hi][colB] = (__bf16)hv;
            }
        }
        if (t == TSTEPS - 1) {   // uniform branch, once
            #pragma unroll
            for (int ci = 0; ci < 4; ++ci) {
                const int colB = (wv * 4 + ci) * 16 + lm;
                #pragma unroll
                for (int g = 0; g < 8; ++g)
                    hlast[(size_t)(rowBase + g + 8 * hi) * HDIM + colB] = ev[ci][g];
            }
        }
        __syncthreads();  // b2: hbuf now holds complete h_t

        // ---- stage 5: output GEMM, one 16x16 logit tile per wave, K=1024 ----
        v8f co = {};
        #pragma unroll 4
        for (int kb = 0; kb < HDIM / 32; ++kb) {
            FragB16 a, b;
            const int kA = kb * 32 + hi * 16;
            a.q[0] = lds_b128(&hbuf[lm][kA]);
            a.q[1] = lds_b128(&hbuf[lm][kA + 8]);
            const unsigned kB = (unsigned)(kb * 32 + hi * 16) * 2;
            b.q[0] = ldg_b128(Wog + roo + kB);
            b.q[1] = ldg_b128(Wog + roo + kB + 16);
            co = __builtin_amdgcn_wmma_f32_16x16x32_bf16(false, a.v, false, b.v, (short)0, co, false, false);
        }
        {
            const int colO = wv * 16 + lm;
            #pragma unroll
            for (int g = 0; g < 8; ++g)
                lg[g + 8 * hi][colO] = co[g] + bo;
        }
        __syncthreads();  // b3: logits complete

        // ---- stage 7: softmax, wave wv owns sample row m = wv ----
        {
            float e[8];
            float mx = -3.0e38f;
            #pragma unroll
            for (int j = 0; j < 8; ++j) {
                e[j] = lg[wv][j * 32 + l];
                mx = fmaxf(mx, e[j]);
            }
            for (int off = 16; off > 0; off >>= 1)
                mx = fmaxf(mx, __shfl_xor(mx, off, 32));
            float s = 0.0f;
            #pragma unroll
            for (int j = 0; j < 8; ++j) {
                e[j] = __expf(e[j] - mx);
                s += e[j];
            }
            for (int off = 16; off > 0; off >>= 1)
                s += __shfl_xor(s, off, 32);
            const float inv = 1.0f / s;
            const size_t base = ((size_t)(rowBase + wv) * TSTEPS + t) * OUTD;
            #pragma unroll
            for (int j = 0; j < 8; ++j)
                y[base + j * 32 + l] = e[j] * inv;
        }
        // No barrier needed here: the next iteration's first LDS writes sit
        // behind b1/b2, which order them after every wave's softmax reads.
    }
}

// ---------------------------------------------------------------------------
extern "C" void kernel_launch(void* const* d_in, const int* in_sizes, int n_in,
                              void* d_out, int out_size, void* d_ws, size_t ws_size,
                              hipStream_t stream) {
    const int*   x    = (const int*)  d_in[0];
    const float* E    = (const float*)d_in[1];
    const float* Whh  = (const float*)d_in[2];
    const float* bh   = (const float*)d_in[3];
    const float* Wout = (const float*)d_in[4];
    const float* bout = (const float*)d_in[5];

    float* y     = (float*)d_out;                                   // (N,T,OUT)
    float* hlast = y + (size_t)NB * TSTEPS * OUTD;                  // (N,H)

    __bf16* Wpk    = (__bf16*)d_ws;                                 // 2 MB
    __bf16* WoutPk = (__bf16*)((char*)d_ws + (size_t)HDIM * HDIM * 2); // 0.5 MB

    prep_weights<<<(HDIM * HDIM + 255) / 256, 256, 0, stream>>>(Whh, Wout, Wpk, WoutPk);
    rnn_fused<<<NB / 16, 512, 0, stream>>>(x, E, bh, bout, Wpk, WoutPk, y, hlast);
}
